// GraphAttention_17248588660853
// MI455X (gfx1250) — matrix operand order repaired and checked by reading, not compiled, not run on hardware
//
#include <hip/hip_runtime.h>
#include <hip/hip_bf16.h>

// ---------------------------------------------------------------------------
// GAT forward, fused flash-attention style for MI455X (gfx1250, wave32).
// N=8192, F_IN=512, F_OUT=256.
//   k_gemm_h : h = x@W        (f32 WMMA 16x16x4), also writes hT as bf16 (transposed)
//   k_f12    : f1 = h@a1, f2 = h@a2 (wave reductions)
//   k_attn   : online-softmax over masked LeakyReLU scores, P@h via bf16 WMMA
//
// Traffic budget (23.3 TB/s HBM): adjacency stream 268 MB dominates; x/W/h/hT/
// f1/f2 all L2-resident (192 MB). Attention matrix is never materialized.
// ---------------------------------------------------------------------------

typedef __attribute__((ext_vector_type(2)))  float  v2f;
typedef __attribute__((ext_vector_type(4)))  float  v4f;
typedef __attribute__((ext_vector_type(8)))  float  v8f;
typedef __attribute__((ext_vector_type(4)))  int    v4i;
typedef __attribute__((ext_vector_type(16))) __bf16 v16bf;

#define N_NODES 8192
#define F_IN    512
#define F_OUT   256
#define NEG_INF (-9e15f)

__device__ __forceinline__ unsigned short f2bf_bits(float f) {
    unsigned u = __builtin_bit_cast(unsigned, f);
    unsigned r = u + 0x7FFFu + ((u >> 16) & 1u);   // round-to-nearest-even
    return (unsigned short)(r >> 16);
}
__device__ __forceinline__ __bf16 f2bf(float f) {
    unsigned short b = f2bf_bits(f);
    return __builtin_bit_cast(__bf16, b);
}

// ---------------------------------------------------------------------------
// Kernel 1: h = x @ W, f32 WMMA 16x16x4. One wave: 16 rows x 64 cols (4 tiles,
// 32 acc VGPRs -> no spills). 512 row-blocks x 4 col-groups = 2048 waves.
// A fragment (16x4 f32): lane l: M = l%16, VGPR r holds K = 2*(l/16) + r.
// B fragment (4x16 f32): lane l: N = l%16, VGPR r holds K = 2*(l/16) + r.
// C/D (16x16 f32):       lane l: N = l%16, VGPR r holds M = r + 8*(l/16).
// ---------------------------------------------------------------------------
__global__ void __launch_bounds__(256)
k_gemm_h(const float* __restrict__ x, const float* __restrict__ W,
         float* __restrict__ h, unsigned short* __restrict__ hT) {
    const int lane = threadIdx.x & 31;
    const int wave = threadIdx.x >> 5;
    const int wgid = blockIdx.x * (blockDim.x >> 5) + wave;   // 0..2047
    const int rb   = wgid >> 2;          // row block 0..511
    const int cg   = wgid & 3;           // column group 0..3 (64 cols each)
    const int i0   = rb * 16;
    const int n0   = cg * 64;
    const int hi   = lane >> 4;          // 0 or 1
    const int lid  = lane & 15;

    v8f acc[4] = {};

    const float* xrow = x + (size_t)(i0 + lid) * F_IN + 2 * hi;
    for (int k = 0; k < F_IN; k += 4) {
        v2f a = *(const v2f*)(xrow + k);
        const float* wb = W + (size_t)(k + 2 * hi) * F_OUT + n0 + lid;
#pragma unroll
        for (int t = 0; t < 4; ++t) {
            v2f b;
            b.x = wb[t * 16];              // K = k + 2*hi + 0
            b.y = wb[F_OUT + t * 16];      // K = k + 2*hi + 1
            acc[t] = __builtin_amdgcn_wmma_f32_16x16x4_f32(
                false, a, false, b, (short)0, acc[t], false, false);
        }
    }

#pragma unroll
    for (int t = 0; t < 4; ++t) {
        const int col = n0 + t * 16 + lid;
#pragma unroll
        for (int r = 0; r < 8; ++r) {
            const int row = i0 + r + 8 * hi;
            float v = acc[t][r];
            h[(size_t)row * F_OUT + col]    = v;
            hT[(size_t)col * N_NODES + row] = f2bf_bits(v);   // transposed bf16
        }
    }
}

// ---------------------------------------------------------------------------
// Kernel 2: f1 = h@a1, f2 = h@a2. One wave per row (wave32 shuffle reduction).
// ---------------------------------------------------------------------------
__global__ void __launch_bounds__(256)
k_f12(const float* __restrict__ h, const float* __restrict__ a1,
      const float* __restrict__ a2,
      float* __restrict__ f1, float* __restrict__ f2) {
    const int lane = threadIdx.x & 31;
    const int wave = threadIdx.x >> 5;
    const int row  = blockIdx.x * (blockDim.x >> 5) + wave;
    const float* hr = h + (size_t)row * F_OUT;
    float s1 = 0.f, s2 = 0.f;
#pragma unroll
    for (int c = lane; c < F_OUT; c += 32) {
        float v = hr[c];
        s1 += v * a1[c];
        s2 += v * a2[c];
    }
#pragma unroll
    for (int off = 16; off > 0; off >>= 1) {
        s1 += __shfl_xor(s1, off, 32);
        s2 += __shfl_xor(s2, off, 32);
    }
    if (lane == 0) { f1[row] = s1; f2[row] = s2; }
}

// ---------------------------------------------------------------------------
// Kernel 3: fused masked-softmax attention. One wave per 16 output rows,
// full 256-wide accumulator (16 tiles, 128 VGPRs) so the 268 MB adjacency
// matrix is streamed exactly once. 128-thread blocks (1 wave/SIMD) to give
// the register allocator the whole VGPR file -> no spills.
// A frag (16x32 bf16): lane l: M=l%16; lanes0-15 hold K=0..15 (v0..7),
//   lanes16-31 hold K=16..31: K = 16*(l/16)+q, same order as B. With s[] ordered as
//   [chunkA(8), chunkB(8)], element q of v16bf == s[q].
// B frag (32x16 bf16): lane l: N=l%16; all 8 VGPRs hold K = 16*(l/16)+0..15,
//   contiguous in K -> one 32-byte load from transposed hT.
// ---------------------------------------------------------------------------
__global__ void __launch_bounds__(128)
k_attn(const int* __restrict__ adj, const unsigned short* __restrict__ hT,
       const float* __restrict__ f1, const float* __restrict__ f2,
       float* __restrict__ out) {
    const int lane = threadIdx.x & 31;
    const int wave = threadIdx.x >> 5;
    const int rb   = blockIdx.x * (blockDim.x >> 5) + wave;   // 0..511
    const int i0   = rb * 16;
    const int hi   = lane >> 4;
    const int lid  = lane & 15;

    v8f acc[16] = {};
    float m    = NEG_INF;   // running row max (matches reference mask value)
    float lsum = 0.f;       // running row sum

    const float f1r  = f1[i0 + lid];
    const int*  arow = adj + (size_t)(i0 + lid) * N_NODES;

    for (int jb = 0; jb < N_NODES; jb += 32) {
        const int base1 = jb + 16 * hi;     // chunk A start (8 cols)
        const int base2 = base1 + 8;        // chunk B start (8 cols)

        // prefetch next adjacency tile while we compute
        __builtin_prefetch(arow + base1 + 32, 0, 0);

        // one-shot 268 MB stream: keep it out of the hot L2 working set
        v4i A0 = __builtin_nontemporal_load((const v4i*)(arow + base1));
        v4i A1 = __builtin_nontemporal_load((const v4i*)(arow + base1 + 4));
        v4i A2 = __builtin_nontemporal_load((const v4i*)(arow + base2));
        v4i A3 = __builtin_nontemporal_load((const v4i*)(arow + base2 + 4));
        v4f G0 = *(const v4f*)(f2 + base1);
        v4f G1 = *(const v4f*)(f2 + base1 + 4);
        v4f G2 = *(const v4f*)(f2 + base2);
        v4f G3 = *(const v4f*)(f2 + base2 + 4);

        const int   av[16] = {A0.x, A0.y, A0.z, A0.w, A1.x, A1.y, A1.z, A1.w,
                              A2.x, A2.y, A2.z, A2.w, A3.x, A3.y, A3.z, A3.w};
        const float fv[16] = {G0.x, G0.y, G0.z, G0.w, G1.x, G1.y, G1.z, G1.w,
                              G2.x, G2.y, G2.z, G2.w, G3.x, G3.y, G3.z, G3.w};

        float s[16];
        float mloc = NEG_INF;
#pragma unroll
        for (int q = 0; q < 16; ++q) {
            float e = f1r + fv[q];
            e = (e > 0.f) ? e : 0.2f * e;            // LeakyReLU(0.2)
            e = (av[q] > 0) ? e : NEG_INF;           // adjacency mask
            s[q] = e;
            mloc = fmaxf(mloc, e);
        }
        // row max across the lane pair (l, l^16) holding the same M
        const float mrow = fmaxf(mloc, __shfl_xor(mloc, 16, 32));
        const float mnew = fmaxf(m, mrow);
        const float scale = __expf(m - mnew);        // finite: m,mnew >= -9e15
        m = mnew;

        float ls = 0.f;
        v16bf afrag;
#pragma unroll
        for (int q = 0; q < 16; ++q) {
            float p = __expf(s[q] - mnew);
            ls += p;
            afrag[q] = f2bf(p);
        }
        ls += __shfl_xor(ls, 16, 32);
        lsum = lsum * scale + ls;

        // per-output-row rescale factors (row = r + 8*hi lives at lane == row)
        float sc[8];
#pragma unroll
        for (int r = 0; r < 8; ++r) sc[r] = __shfl(scale, r + 8 * hi, 32);
#pragma unroll
        for (int t = 0; t < 16; ++t)
#pragma unroll
            for (int r = 0; r < 8; ++r) acc[t][r] *= sc[r];

        // 16 bf16 WMMAs: P(16x32) @ h[jb:jb+32, :]
        const unsigned short* hTb = hT + jb + 16 * hi;
#pragma unroll
        for (int t = 0; t < 16; ++t) {
            const unsigned short* p = hTb + (size_t)(t * 16 + lid) * N_NODES;
            v16bf bfrag = *(const v16bf*)p;          // 32 contiguous bytes
            acc[t] = __builtin_amdgcn_wmma_f32_16x16x32_bf16(
                false, afrag, false, bfrag, (short)0, acc[t], false, false);
        }
    }

    // normalize, ELU, store
    float il[8];
#pragma unroll
    for (int r = 0; r < 8; ++r) {
        float lr = __shfl(lsum, r + 8 * hi, 32);
        il[r] = 1.0f / lr;
    }
#pragma unroll
    for (int t = 0; t < 16; ++t) {
        const int col = t * 16 + lid;
#pragma unroll
        for (int r = 0; r < 8; ++r) {
            float v = acc[t][r] * il[r];
            v = (v > 0.f) ? v : (__expf(v) - 1.f);   // ELU
            __builtin_nontemporal_store(v, out + (size_t)(i0 + r + 8 * hi) * F_OUT + col);
        }
    }
}

// ---------------------------------------------------------------------------
// Launch. Workspace layout:
//   h   f32 [8192][256]          8,388,608 B
//   hT  bf16[256][8192]          4,194,304 B   (transposed)
//   f1  f32 [8192]                  32,768 B
//   f2  f32 [8192]                  32,768 B
// ---------------------------------------------------------------------------
extern "C" void kernel_launch(void* const* d_in, const int* in_sizes, int n_in,
                              void* d_out, int out_size, void* d_ws, size_t ws_size,
                              hipStream_t stream) {
    const float* x   = (const float*)d_in[0];
    const int*   adj = (const int*)  d_in[1];
    const float* W   = (const float*)d_in[2];
    const float* a1  = (const float*)d_in[3];
    const float* a2  = (const float*)d_in[4];
    float* out = (float*)d_out;

    char* ws = (char*)d_ws;
    float*          h  = (float*)ws;
    unsigned short* hT = (unsigned short*)(ws + (size_t)N_NODES * F_OUT * 4);
    float*          f1 = (float*)(ws + (size_t)N_NODES * F_OUT * 4
                                     + (size_t)F_OUT * N_NODES * 2);
    float*          f2 = f1 + N_NODES;

    // 2048 waves: 512 row-blocks x 4 column-groups, 8 waves per 256-thread block
    k_gemm_h<<<256,  256, 0, stream>>>(x, W, h, hT);
    // 8192 rows, one wave each
    k_f12  <<<1024,  256, 0, stream>>>(h, a1, a2, f1, f2);
    // 512 row-blocks, 4 waves per 128-thread block (full VGPR budget per wave)
    k_attn <<<128,   128, 0, stream>>>(adj, hT, f1, f2, out);
}
